// GAT_86560770883965
// MI455X (gfx1250) — matrix operand order repaired and checked by reading, not compiled, not run on hardware
//
#include <hip/hip_runtime.h>
#include <hip/hip_bf16.h>

typedef __attribute__((ext_vector_type(16))) _Float16 v16h;
typedef __attribute__((ext_vector_type(8)))  _Float16 v8h;
typedef __attribute__((ext_vector_type(8)))  float    v8f;

#define NEG_SLOPE 0.2f
#define ENC_NEG_INF 0x007FFFFFu   /* order-preserving encoding of -inf */

__device__ __forceinline__ unsigned enc_f32(float f) {
    unsigned u = __float_as_uint(f);
    return (u & 0x80000000u) ? ~u : (u | 0x80000000u);
}
__device__ __forceinline__ float dec_f32(unsigned u) {
    return __uint_as_float((u & 0x80000000u) ? (u & 0x7fffffffu) : ~u);
}

// ---------------- precision convert ----------------
__global__ void k_f32_to_f16(const float* __restrict__ src,
                             _Float16* __restrict__ dst, int n) {
    int i = blockIdx.x * blockDim.x + threadIdx.x;
    if (i < n) dst[i] = (_Float16)src[i];
}

// ---------------- pack W into CDNA5 WMMA B-fragment layout ----------------
// B fragment (16-bit, 32x16): lane<16 -> N=lane,    elem e<8 -> K = kt*32 + e,     e>=8 -> K = kt*32 + 16 + (e-8)
//                             lane>=16-> N=lane-16, elem e<8 -> K = kt*32 + 8 + e, e>=8 -> K = kt*32 + 24 + (e-8)
// fragment id = nt*nKT + kt, each fragment = 32 lanes * 16 halves, contiguous.
__global__ void k_pack_b(const float* __restrict__ W, _Float16* __restrict__ out,
                         int K, int Ncols) {
    int o = blockIdx.x * blockDim.x + threadIdx.x;
    int total = K * Ncols;
    if (o >= total) return;
    int nKT  = K >> 5;
    int frag = o >> 9;          // / (32*16)
    int rem  = o & 511;
    int lane = rem >> 4;
    int e    = rem & 15;
    int kt = frag % nKT;
    int nt = frag / nKT;
    int k = kt * 32 + ((lane < 16) ? 0 : 8) + ((e < 8) ? e : 16 + (e - 8));
    int c = nt * 16 + (lane & 15);
    out[o] = (_Float16)W[(size_t)k * Ncols + c];
}

// ---------------- WMMA GEMM: C[nrows,NCOLS] = A[nrows,K] * Bpacked ----------------
// One wave computes a 16-row x NCOLS tile. A is f16 row-major; A fragment layout:
// lane row = lane&15; elems 0..7 -> K = kt*32 + (lane<16?0:8) + i
//                     elems 8..15-> K = kt*32 + 16 + (lane<16?0:8) + i
template <int K, int NCOLS>
__global__ __launch_bounds__(64) void k_wmma_gemm(const _Float16* __restrict__ A,
                                                  const _Float16* __restrict__ Bp,
                                                  float* __restrict__ C, int nrows) {
    constexpr int KT = K / 32;
    constexpr int NT = NCOLS / 16;
    int wave = (int)((blockIdx.x * blockDim.x + threadIdx.x) >> 5);
    int lane = threadIdx.x & 31;
    int row0 = wave << 4;
    if (row0 >= nrows) return;

    const _Float16* arow = A + (size_t)(row0 + (lane & 15)) * K;
    const int koff = (lane < 16) ? 0 : 8;

    v8f acc[NT] = {};

#pragma unroll
    for (int kt = 0; kt < KT; ++kt) {
        v8h lo = *(const v8h*)(arow + kt * 32 + koff);
        v8h hi = *(const v8h*)(arow + kt * 32 + 16 + koff);
        v16h a = __builtin_shufflevector(lo, hi, 0, 1, 2, 3, 4, 5, 6, 7,
                                                  8, 9, 10, 11, 12, 13, 14, 15);
#pragma unroll
        for (int nt = 0; nt < NT; ++nt) {
            v16h b = *(const v16h*)(Bp + ((size_t)(nt * KT + kt) * 32 + lane) * 16);
            acc[nt] = __builtin_amdgcn_wmma_f32_16x16x32_f16(
                false, a, false, b, (short)0, acc[nt], false, false);
        }
    }

    // C/D layout: VGPR v, lanes 0-15 -> M=v, N=lane ; lanes 16-31 -> M=v+8, N=lane-16
    int rbase = row0 + ((lane & 16) ? 8 : 0);
    int cbase = lane & 15;
#pragma unroll
    for (int nt = 0; nt < NT; ++nt) {
#pragma unroll
        for (int v = 0; v < 8; ++v) {
            C[(size_t)(rbase + v) * NCOLS + nt * 16 + cbase] = acc[nt][v];
        }
    }
}

// ---------------- layer-1 attention logits + scratch init ----------------
__global__ void k_al1(const float* __restrict__ h, const float* __restrict__ a_s,
                      const float* __restrict__ a_d, float* __restrict__ al_s,
                      float* __restrict__ al_d, unsigned* __restrict__ m,
                      float* __restrict__ den, float* __restrict__ acc, int n) {
    int t = blockIdx.x * blockDim.x + threadIdx.x;
    if (t >= n * 8) return;
    int node = t >> 3, hd = t & 7;
    const float* hp = h + (size_t)node * 128 + hd * 16;
    float s = 0.f, d = 0.f;
#pragma unroll
    for (int i = 0; i < 16; ++i) {
        float v = hp[i];
        s += v * a_s[hd * 16 + i];
        d += v * a_d[hd * 16 + i];
    }
    al_s[t] = s; al_d[t] = d;
    m[t] = ENC_NEG_INF; den[t] = 0.f;
    float* ap = acc + (size_t)node * 128 + hd * 16;
#pragma unroll
    for (int i = 0; i < 16; ++i) ap[i] = 0.f;
}

// ---------------- layer-1 edge pass 1: segment max ----------------
__global__ void k_e1_max(const int* __restrict__ ei, int E, int n,
                         const float* __restrict__ al_s, const float* __restrict__ al_d,
                         unsigned* __restrict__ m) {
    int t = blockIdx.x * blockDim.x + threadIdx.x;
    int ET = E + n;
    if (t >= ET * 8) return;
    int eid = t >> 3, hd = t & 7;
    int src, dst;
    if (eid < E) { src = ei[eid]; dst = ei[E + eid]; } else { src = dst = eid - E; }
    float e = al_s[src * 8 + hd] + al_d[dst * 8 + hd];
    e = (e > 0.f) ? e : NEG_SLOPE * e;
    atomicMax(&m[dst * 8 + hd], enc_f32(e));
}

// ---------------- layer-1 edge pass 2: exp + weighted scatter-add ----------------
__global__ void k_e1_acc(const int* __restrict__ ei, int E, int n,
                         const float* __restrict__ al_s, const float* __restrict__ al_d,
                         const unsigned* __restrict__ m, const float* __restrict__ h,
                         float* __restrict__ den, float* __restrict__ acc) {
    int t = blockIdx.x * blockDim.x + threadIdx.x;
    int ET = E + n;
    if (t >= ET * 8) return;
    int eid = t >> 3, hd = t & 7;
    int src, dst;
    if (eid < E) { src = ei[eid]; dst = ei[E + eid]; } else { src = dst = eid - E; }
    float e = al_s[src * 8 + hd] + al_d[dst * 8 + hd];
    e = (e > 0.f) ? e : NEG_SLOPE * e;
    float ex = expf(e - dec_f32(m[dst * 8 + hd]));
    atomicAdd(&den[dst * 8 + hd], ex);
    const float4* hs = (const float4*)(h + (size_t)src * 128 + hd * 16);
    float* ad = acc + (size_t)dst * 128 + hd * 16;
#pragma unroll
    for (int q = 0; q < 4; ++q) {
        float4 v = hs[q];
        atomicAdd(&ad[q * 4 + 0], ex * v.x);
        atomicAdd(&ad[q * 4 + 1], ex * v.y);
        atomicAdd(&ad[q * 4 + 2], ex * v.z);
        atomicAdd(&ad[q * 4 + 3], ex * v.w);
    }
}

// ---------------- finalize layer 1: normalize + bias + ELU -> f16 ----------------
__global__ void k_fin1(const float* __restrict__ acc, const float* __restrict__ den,
                       const float* __restrict__ b, _Float16* __restrict__ out, int n) {
    int t = blockIdx.x * blockDim.x + threadIdx.x;
    if (t >= n * 128) return;
    int node = t >> 7, c = t & 127;
    float v = acc[t] / (den[node * 8 + (c >> 4)] + 1e-16f) + b[c];
    v = (v > 0.f) ? v : expm1f(v);
    out[t] = (_Float16)v;
}

// ---------------- layer-2 attention logits + scratch init ----------------
__global__ void k_al2(const float* __restrict__ h2, const float* __restrict__ a_s,
                      const float* __restrict__ a_d, float* __restrict__ al_s,
                      float* __restrict__ al_d, unsigned* __restrict__ m,
                      float* __restrict__ den, float* __restrict__ acc, int n) {
    int node = blockIdx.x * blockDim.x + threadIdx.x;
    if (node >= n) return;
    const float* hp = h2 + (size_t)node * 64;
    float s = 0.f, d = 0.f;
#pragma unroll
    for (int i = 0; i < 64; ++i) { float v = hp[i]; s += v * a_s[i]; d += v * a_d[i]; }
    al_s[node] = s; al_d[node] = d;
    m[node] = ENC_NEG_INF; den[node] = 0.f;
    float* ap = acc + (size_t)node * 64;
#pragma unroll
    for (int i = 0; i < 64; ++i) ap[i] = 0.f;
}

__global__ void k_e2_max(const int* __restrict__ ei, int E, int n,
                         const float* __restrict__ al_s, const float* __restrict__ al_d,
                         unsigned* __restrict__ m) {
    int eid = blockIdx.x * blockDim.x + threadIdx.x;
    int ET = E + n;
    if (eid >= ET) return;
    int src, dst;
    if (eid < E) { src = ei[eid]; dst = ei[E + eid]; } else { src = dst = eid - E; }
    float e = al_s[src] + al_d[dst];
    e = (e > 0.f) ? e : NEG_SLOPE * e;
    atomicMax(&m[dst], enc_f32(e));
}

// one thread per (edge, 16-column chunk)
__global__ void k_e2_acc(const int* __restrict__ ei, int E, int n,
                         const float* __restrict__ al_s, const float* __restrict__ al_d,
                         const unsigned* __restrict__ m, const float* __restrict__ h2,
                         float* __restrict__ den, float* __restrict__ acc) {
    int t = blockIdx.x * blockDim.x + threadIdx.x;
    int ET = E + n;
    if (t >= ET * 4) return;
    int eid = t >> 2, q = t & 3;
    int src, dst;
    if (eid < E) { src = ei[eid]; dst = ei[E + eid]; } else { src = dst = eid - E; }
    float e = al_s[src] + al_d[dst];
    e = (e > 0.f) ? e : NEG_SLOPE * e;
    float ex = expf(e - dec_f32(m[dst]));
    if (q == 0) atomicAdd(&den[dst], ex);
    const float4* hs = (const float4*)(h2 + (size_t)src * 64 + q * 16);
    float* ad = acc + (size_t)dst * 64 + q * 16;
#pragma unroll
    for (int j = 0; j < 4; ++j) {
        float4 v = hs[j];
        atomicAdd(&ad[j * 4 + 0], ex * v.x);
        atomicAdd(&ad[j * 4 + 1], ex * v.y);
        atomicAdd(&ad[j * 4 + 2], ex * v.z);
        atomicAdd(&ad[j * 4 + 3], ex * v.w);
    }
}

// ---------------- final: normalize + bias + log_softmax (wave per node) ----------------
__global__ __launch_bounds__(256) void k_lsm(const float* __restrict__ acc,
                                             const float* __restrict__ den,
                                             const float* __restrict__ b,
                                             float* __restrict__ out, int n) {
    int wave = (int)((blockIdx.x * blockDim.x + threadIdx.x) >> 5);
    int lane = threadIdx.x & 31;
    if (wave >= n) return;
    const float* ap = acc + (size_t)wave * 64;
    float dn = den[wave] + 1e-16f;
    float v0 = ap[lane] / dn + b[lane];
    float v1 = ap[lane + 32] / dn + b[lane + 32];
    float mx = fmaxf(v0, v1);
#pragma unroll
    for (int off = 16; off; off >>= 1) mx = fmaxf(mx, __shfl_xor(mx, off, 32));
    float s = expf(v0 - mx) + expf(v1 - mx);
#pragma unroll
    for (int off = 16; off; off >>= 1) s += __shfl_xor(s, off, 32);
    float lse = logf(s) + mx;
    out[(size_t)wave * 64 + lane] = v0 - lse;
    out[(size_t)wave * 64 + lane + 32] = v1 - lse;
}

static inline int cdiv(long long a, long long b) { return (int)((a + b - 1) / b); }

extern "C" void kernel_launch(void* const* d_in, const int* in_sizes, int n_in,
                              void* d_out, int out_size, void* d_ws, size_t ws_size,
                              hipStream_t stream) {
    const float* x   = (const float*)d_in[0];
    const int*   ei  = (const int*)d_in[1];
    const float* W1  = (const float*)d_in[2];
    const float* as1 = (const float*)d_in[3];
    const float* ad1 = (const float*)d_in[4];
    const float* b1  = (const float*)d_in[5];
    const float* W2  = (const float*)d_in[6];
    const float* as2 = (const float*)d_in[7];
    const float* ad2 = (const float*)d_in[8];
    const float* b2  = (const float*)d_in[9];
    float* out = (float*)d_out;

    const int IN = 256, H1 = 128, OC = 64;
    const int Nn = in_sizes[0] / IN;
    const int E  = in_sizes[1] / 2;
    const long long ET = (long long)E + Nn;

    char* ws = (char*)d_ws;
    size_t off = 0;
    auto alloc = [&](size_t bytes) {
        size_t o = off;
        off += (bytes + 255) & ~(size_t)255;
        return o;
    };
    size_t o_xh   = alloc((size_t)Nn * IN * 2);   // x f16; later reused (h1act + acc2)
    size_t o_w1p  = alloc((size_t)IN * H1 * 2);
    size_t o_w2p  = alloc((size_t)H1 * OC * 2);
    size_t o_h1   = alloc((size_t)Nn * H1 * 4);   // layer1 pre-attention features
    size_t o_acc1 = alloc((size_t)Nn * H1 * 4);
    size_t o_als1 = alloc((size_t)Nn * 8 * 4);
    size_t o_ald1 = alloc((size_t)Nn * 8 * 4);
    size_t o_m1   = alloc((size_t)Nn * 8 * 4);
    size_t o_den1 = alloc((size_t)Nn * 8 * 4);
    size_t o_h2   = alloc((size_t)Nn * OC * 4);
    size_t o_als2 = alloc((size_t)Nn * 4);
    size_t o_ald2 = alloc((size_t)Nn * 4);
    size_t o_m2   = alloc((size_t)Nn * 4);
    size_t o_den2 = alloc((size_t)Nn * 4);
    // reuse the (dead after GEMM1) x_h region:
    size_t o_h1a  = o_xh;                                                   // N*H1 f16
    size_t o_acc2 = o_xh + (((size_t)Nn * H1 * 2 + 255) & ~(size_t)255);    // N*OC f32

    _Float16* xh   = (_Float16*)(ws + o_xh);
    _Float16* w1p  = (_Float16*)(ws + o_w1p);
    _Float16* w2p  = (_Float16*)(ws + o_w2p);
    float*    h1   = (float*)(ws + o_h1);
    float*    acc1 = (float*)(ws + o_acc1);
    float*    als1 = (float*)(ws + o_als1);
    float*    ald1 = (float*)(ws + o_ald1);
    unsigned* m1   = (unsigned*)(ws + o_m1);
    float*    den1 = (float*)(ws + o_den1);
    _Float16* h1a  = (_Float16*)(ws + o_h1a);
    float*    h2   = (float*)(ws + o_h2);
    float*    als2 = (float*)(ws + o_als2);
    float*    ald2 = (float*)(ws + o_ald2);
    unsigned* m2   = (unsigned*)(ws + o_m2);
    float*    den2 = (float*)(ws + o_den2);
    float*    acc2 = (float*)(ws + o_acc2);

    const int B = 256;
    int tiles = cdiv(Nn, 16);

    // ---- layer 1 ----
    k_f32_to_f16<<<cdiv((long long)Nn * IN, B), B, 0, stream>>>(x, xh, Nn * IN);
    k_pack_b<<<cdiv(IN * H1, B), B, 0, stream>>>(W1, w1p, IN, H1);
    k_pack_b<<<cdiv(H1 * OC, B), B, 0, stream>>>(W2, w2p, H1, OC);
    k_wmma_gemm<256, 128><<<cdiv(tiles, 2), 64, 0, stream>>>(xh, w1p, h1, Nn);
    k_al1<<<cdiv((long long)Nn * 8, B), B, 0, stream>>>(h1, as1, ad1, als1, ald1, m1, den1, acc1, Nn);
    k_e1_max<<<cdiv(ET * 8, B), B, 0, stream>>>(ei, E, Nn, als1, ald1, m1);
    k_e1_acc<<<cdiv(ET * 8, B), B, 0, stream>>>(ei, E, Nn, als1, ald1, m1, h1, den1, acc1);
    k_fin1<<<cdiv((long long)Nn * H1, B), B, 0, stream>>>(acc1, den1, b1, h1a, Nn);

    // ---- layer 2 ----
    k_wmma_gemm<128, 64><<<cdiv(tiles, 2), 64, 0, stream>>>(h1a, w2p, h2, Nn);
    k_al2<<<cdiv(Nn, B), B, 0, stream>>>(h2, as2, ad2, als2, ald2, m2, den2, acc2, Nn);
    k_e2_max<<<cdiv(ET, B), B, 0, stream>>>(ei, E, Nn, als2, ald2, m2);
    k_e2_acc<<<cdiv(ET * 4, B), B, 0, stream>>>(ei, E, Nn, als2, ald2, m2, h2, den2, acc2);
    k_lsm<<<cdiv((long long)Nn * 32, B), B, 0, stream>>>(acc2, den2, b2, out, Nn);
}
